// ManifoldAttention_68891275427894
// MI455X (gfx1250) — compile-verified, matrix operand order checked by reading, not run on hardware
//
#include <hip/hip_runtime.h>

#define B_   2
#define N_   2048
#define D_   512
#define H_   8
#define F_   8
#define HD_  64
#define AUG_ 96   // 64 head dims + 8 cos + 8 sin + 16 zero pad (3 chunks of 32)

typedef __attribute__((ext_vector_type(16))) _Float16 v16h;
typedef __attribute__((ext_vector_type(8)))  _Float16 v8h;
typedef __attribute__((ext_vector_type(8)))  float    v8f;

// ---------------------------------------------------------------------------
// WMMA helpers
// ---------------------------------------------------------------------------
__device__ __forceinline__ v8f wmma_f16(v16h a, v16h b, v8f c) {
  return __builtin_amdgcn_wmma_f32_16x16x32_f16(
      /*neg_a=*/false, a, /*neg_b=*/false, b,
      /*c_mod=*/(short)0, c, /*reuse_a=*/false, /*reuse_b=*/false);
}

// Load a 16x32 f16 fragment (A-layout; B-layout is identical with rows=cols)
// from row-major storage with leading dimension `ld` (halfs).
// ISA 7.12.2 16-bit layout: lane<16: M=lane, K in {0..7}(v0..3),{16..23}(v4..7)
//                           lane>=16: M=lane-16, K in {8..15},{24..31}
__device__ __forceinline__ v16h load_frag(const _Float16* __restrict__ p,
                                          int ld, int row0, int k0) {
  int lane = threadIdx.x & 31;
  int r    = lane & 15;
  int koff = (lane & 16) ? 8 : 0;
  const _Float16* q = p + (size_t)(row0 + r) * (size_t)ld + k0 + koff;
  v8h lo = *(const v8h*)q;
  v8h hi = *(const v8h*)(q + 16);
  v16h out;
#pragma unroll
  for (int i = 0; i < 8; ++i) { out[i] = lo[i]; out[i + 8] = hi[i]; }
  return out;
}

// Same pattern for an LDS-resident 16x32 tile (ld = 32 halfs).
__device__ __forceinline__ v16h load_frag_lds(const _Float16* p) {
  int lane = threadIdx.x & 31;
  int r    = lane & 15;
  int koff = (lane & 16) ? 8 : 0;
  const _Float16* q = p + r * 32 + koff;
  v8h lo = *(const v8h*)q;
  v8h hi = *(const v8h*)(q + 16);
  v16h out;
#pragma unroll
  for (int i = 0; i < 8; ++i) { out[i] = lo[i]; out[i + 8] = hi[i]; }
  return out;
}

// ---------------------------------------------------------------------------
// 1) fp32 -> f16 convert
// ---------------------------------------------------------------------------
__global__ void cvt_f32_f16(const float* __restrict__ src,
                            _Float16* __restrict__ dst, int n) {
  int stride = gridDim.x * blockDim.x;
  for (int i = blockIdx.x * blockDim.x + threadIdx.x; i < n; i += stride)
    dst[i] = (_Float16)src[i];
}

// ---------------------------------------------------------------------------
// 2) Fused QKV projection: P = x @ W^T + b, scattered into per-head layouts.
//    blockIdx.z selects Q(0)/K(1)/V(2). One wave = one 32x32 output tile
//    (2 A-frags x 2 B-frags -> 4 WMMA per 32-deep k-chunk; ~1 load per wmma).
//    Q -> Qaug[bh][n][0..63] scaled by 1/sqrt(HD)=1/8
//    K -> Kaug[bh][n][0..63]
//    V -> Vt  [bh][hd][n]    (transposed so PV B-frags load contiguously)
// ---------------------------------------------------------------------------
__global__ __launch_bounds__(128)
void proj_qkv(const _Float16* __restrict__ xh,
              const _Float16* __restrict__ Wqh, const _Float16* __restrict__ Wkh,
              const _Float16* __restrict__ Wvh,
              const float* __restrict__ bq, const float* __restrict__ bk,
              const float* __restrict__ bv,
              _Float16* __restrict__ Qaug, _Float16* __restrict__ Kaug,
              _Float16* __restrict__ Vt) {
  int which = blockIdx.z;
  const _Float16* W;
  const float* bias;
  if (which == 0)      { W = Wqh; bias = bq; }
  else if (which == 1) { W = Wkh; bias = bk; }
  else                 { W = Wvh; bias = bv; }

  int waveId = blockIdx.x * 4 + (threadIdx.x >> 5);
  int mt = waveId >> 4;        // 0..127  (4096 rows / 32)
  int jt = waveId & 15;        // 0..15   (512 cols / 32)

  v8f acc[2][2] = {};
#pragma unroll 2
  for (int k0 = 0; k0 < D_; k0 += 32) {
    v16h a0 = load_frag(xh, D_, mt * 32,      k0);
    v16h a1 = load_frag(xh, D_, mt * 32 + 16, k0);
    v16h b0 = load_frag(W,  D_, jt * 32,      k0);
    v16h b1 = load_frag(W,  D_, jt * 32 + 16, k0);
    acc[0][0] = wmma_f16(a0, b0, acc[0][0]);
    acc[0][1] = wmma_f16(a0, b1, acc[0][1]);
    acc[1][0] = wmma_f16(a1, b0, acc[1][0]);
    acc[1][1] = wmma_f16(a1, b1, acc[1][1]);
  }

  int lane  = threadIdx.x & 31;
  int rhalf = (lane & 16) ? 8 : 0;
  int coll  = lane & 15;
#pragma unroll
  for (int sj = 0; sj < 2; ++sj) {
    int col = jt * 32 + sj * 16 + coll;   // output feature in [0,512)
    int h   = col >> 6;
    int hd  = col & 63;
    float bcol = bias[col];
#pragma unroll
    for (int sm = 0; sm < 2; ++sm) {
      int rbase = mt * 32 + sm * 16 + rhalf;
#pragma unroll
      for (int r = 0; r < 8; ++r) {
        int row = rbase + r;              // global row in [0, B*N)
        int b_  = row >> 11;              // / 2048
        int n   = row & (N_ - 1);
        float v = acc[sm][sj][r] + bcol;
        size_t bh = (size_t)(b_ * H_ + h);
        if (which == 0)
          Qaug[(bh * N_ + n) * AUG_ + hd] = (_Float16)(v * 0.125f);
        else if (which == 1)
          Kaug[(bh * N_ + n) * AUG_ + hd] = (_Float16)v;
        else
          Vt[(bh * HD_ + hd) * (size_t)N_ + n] = (_Float16)v;
      }
    }
  }
}

// ---------------------------------------------------------------------------
// 3) Phase features: Qaug[...,64..79] = [cos*pb_h, sin*pb_h], pad 80..95 = 0
//                    Kaug[...,64..79] = [cos, sin],           pad 80..95 = 0
// ---------------------------------------------------------------------------
__global__ void phase_fill(const float* __restrict__ pc,   // [B,N,F]
                           const float* __restrict__ pb,   // [H,F]
                           _Float16* __restrict__ Qaug,
                           _Float16* __restrict__ Kaug) {
  int idx = blockIdx.x * blockDim.x + threadIdx.x;   // over B*H*N
  if (idx >= B_ * H_ * N_) return;
  int n = idx & (N_ - 1);
  int h = (idx >> 11) & (H_ - 1);
  int b = idx >> 14;
  const float* p = pc + ((size_t)b * N_ + n) * F_;
  size_t base = ((size_t)(b * H_ + h) * N_ + n) * AUG_ + HD_;
  _Float16* q = Qaug + base;
  _Float16* k = Kaug + base;
#pragma unroll
  for (int f = 0; f < F_; ++f) {
    float s, c;
    __sincosf(p[f], &s, &c);
    float w = pb[h * F_ + f];
    q[f]      = (_Float16)(c * w);
    q[F_ + f] = (_Float16)(s * w);
    k[f]      = (_Float16)c;
    k[F_ + f] = (_Float16)s;
  }
#pragma unroll
  for (int i = 2 * F_; i < AUG_ - HD_; ++i) { q[i] = (_Float16)0.f; k[i] = (_Float16)0.f; }
}

// ---------------------------------------------------------------------------
// 4) Flash attention. 1 wave = 32 q-rows (2 q-subtiles); 32 keys/iteration.
//    Per iteration: 12 score WMMAs + 8 PV WMMAs against 20 b128 global loads
//    (K-frags and V-frags each reused across both q-subtiles).
//    scores = Qaug . Kaug^T over 96 dims (3 wmma chunks per 16-key tile).
//    Online softmax in registers; P re-striped via per-wave LDS tiles.
// ---------------------------------------------------------------------------
__global__ __launch_bounds__(128)
void flash_attn(const _Float16* __restrict__ Qaug,
                const _Float16* __restrict__ Kaug,
                const _Float16* __restrict__ Vt,
                _Float16* __restrict__ Ah) {
  __shared__ _Float16 lds[4][2][16 * 32];

  int wave = threadIdx.x >> 5;
  int lane = threadIdx.x & 31;
  int bh   = blockIdx.x >> 4;           // 0..15
  int q128 = blockIdx.x & 15;           // 0..15 (slabs of 128 q rows)
  int b    = bh >> 3;
  int h    = bh & 7;
  int q0   = q128 * 128 + wave * 32;

  const _Float16* Qb = Qaug + (size_t)bh * N_ * AUG_;
  const _Float16* Kb = Kaug + (size_t)bh * N_ * AUG_;
  const _Float16* Vb = Vt   + (size_t)bh * HD_ * N_;

  v16h qf[2][3];
#pragma unroll
  for (int qt = 0; qt < 2; ++qt)
#pragma unroll
    for (int c = 0; c < 3; ++c)
      qf[qt][c] = load_frag(Qb, AUG_, q0 + qt * 16, c * 32);

  v8f O[2][4] = {};
  float mprev[2][8], lsum[2][8];
#pragma unroll
  for (int qt = 0; qt < 2; ++qt)
#pragma unroll
    for (int r = 0; r < 8; ++r) { mprev[qt][r] = -3.0e38f; lsum[qt][r] = 0.f; }

  int rowb = (lane & 16) ? 8 : 0;
  int coll = lane & 15;

  for (int kt = 0; kt < N_; kt += 32) {
    if (kt + 32 < N_)
      __builtin_prefetch(Kb + (size_t)(kt + 32) * AUG_, 0, 1);

    // --- scores: two q-subtiles x two 16-key tiles over 96-wide aug dim ---
    v8f S[2][2] = {};
#pragma unroll
    for (int c = 0; c < 3; ++c) {
      v16h k0f = load_frag(Kb, AUG_, kt,      c * 32);
      v16h k1f = load_frag(Kb, AUG_, kt + 16, c * 32);
#pragma unroll
      for (int qt = 0; qt < 2; ++qt) {
        S[qt][0] = wmma_f16(qf[qt][c], k0f, S[qt][0]);
        S[qt][1] = wmma_f16(qf[qt][c], k1f, S[qt][1]);
      }
    }

    // --- online softmax per q-subtile ---
#pragma unroll
    for (int qt = 0; qt < 2; ++qt) {
      float mloc[8];
#pragma unroll
      for (int r = 0; r < 8; ++r) mloc[r] = fmaxf(S[qt][0][r], S[qt][1][r]);
#pragma unroll
      for (int mask = 1; mask < 16; mask <<= 1)
#pragma unroll
        for (int r = 0; r < 8; ++r)
          mloc[r] = fmaxf(mloc[r], __shfl_xor(mloc[r], mask, 16));

      float scale[8];
#pragma unroll
      for (int r = 0; r < 8; ++r) {
        float mn = fmaxf(mprev[qt][r], mloc[r]);
        scale[r] = __expf(mprev[qt][r] - mn);
        mprev[qt][r] = mn;
      }

      float rs[8];
#pragma unroll
      for (int r = 0; r < 8; ++r) {
        float p0 = __expf(S[qt][0][r] - mprev[qt][r]);
        float p1 = __expf(S[qt][1][r] - mprev[qt][r]);
        S[qt][0][r] = p0; S[qt][1][r] = p1;
        rs[r] = p0 + p1;
      }
#pragma unroll
      for (int mask = 1; mask < 16; mask <<= 1)
#pragma unroll
        for (int r = 0; r < 8; ++r)
          rs[r] += __shfl_xor(rs[r], mask, 16);
#pragma unroll
      for (int r = 0; r < 8; ++r)
        lsum[qt][r] = lsum[qt][r] * scale[r] + rs[r];
#pragma unroll
      for (int c = 0; c < 4; ++c)
#pragma unroll
        for (int r = 0; r < 8; ++r) O[qt][c][r] *= scale[r];

      // re-stripe P (C-layout) into WMMA A-layout via per-wave LDS tile
      _Float16* plds = lds[wave][qt];
#pragma unroll
      for (int r = 0; r < 8; ++r) {
        plds[(rowb + r) * 32 + coll]      = (_Float16)S[qt][0][r];
        plds[(rowb + r) * 32 + 16 + coll] = (_Float16)S[qt][1][r];
      }
    }
    asm volatile("s_wait_dscnt 0" ::: "memory");
    v16h pf0 = load_frag_lds(lds[wave][0]);
    v16h pf1 = load_frag_lds(lds[wave][1]);

    // --- O += P @ V ; each V B-frag feeds both q-subtiles ---
#pragma unroll
    for (int c = 0; c < 4; ++c) {
      v16h vf = load_frag(Vb, N_, c * 16, kt);
      O[0][c] = wmma_f16(pf0, vf, O[0][c]);
      O[1][c] = wmma_f16(pf1, vf, O[1][c]);
    }
  }

  // --- epilogue: normalize and write attended (f16) back in [B,N,D] layout
#pragma unroll
  for (int qt = 0; qt < 2; ++qt)
#pragma unroll
    for (int c = 0; c < 4; ++c)
#pragma unroll
      for (int r = 0; r < 8; ++r) {
        int qrow = q0 + qt * 16 + rowb + r;
        float v = O[qt][c][r] / lsum[qt][r];
        Ah[((size_t)(b * N_ + qrow)) * D_ + h * HD_ + c * 16 + coll] = (_Float16)v;
      }
}

// ---------------------------------------------------------------------------
// 5) Output projection: out = attended @ Wo^T + bo (fp32), 32x32 wave tile.
// ---------------------------------------------------------------------------
__global__ __launch_bounds__(128)
void gemm_out(const _Float16* __restrict__ Ah,
              const _Float16* __restrict__ Woh,
              const float* __restrict__ bo,
              float* __restrict__ out) {
  int waveId = blockIdx.x * 4 + (threadIdx.x >> 5);
  int mt = waveId >> 4;        // 0..127
  int jt = waveId & 15;        // 0..15

  v8f acc[2][2] = {};
#pragma unroll 2
  for (int k0 = 0; k0 < D_; k0 += 32) {
    v16h a0 = load_frag(Ah,  D_, mt * 32,      k0);
    v16h a1 = load_frag(Ah,  D_, mt * 32 + 16, k0);
    v16h b0 = load_frag(Woh, D_, jt * 32,      k0);
    v16h b1 = load_frag(Woh, D_, jt * 32 + 16, k0);
    acc[0][0] = wmma_f16(a0, b0, acc[0][0]);
    acc[0][1] = wmma_f16(a0, b1, acc[0][1]);
    acc[1][0] = wmma_f16(a1, b0, acc[1][0]);
    acc[1][1] = wmma_f16(a1, b1, acc[1][1]);
  }

  int lane  = threadIdx.x & 31;
  int rhalf = (lane & 16) ? 8 : 0;
  int coll  = lane & 15;
#pragma unroll
  for (int sj = 0; sj < 2; ++sj) {
    int col = jt * 32 + sj * 16 + coll;
    float bcol = bo[col];
#pragma unroll
    for (int sm = 0; sm < 2; ++sm) {
      int rbase = mt * 32 + sm * 16 + rhalf;
#pragma unroll
      for (int r = 0; r < 8; ++r)
        out[(size_t)(rbase + r) * D_ + col] = acc[sm][sj][r] + bcol;
    }
  }
}

// ---------------------------------------------------------------------------
// Launch
// ---------------------------------------------------------------------------
extern "C" void kernel_launch(void* const* d_in, const int* in_sizes, int n_in,
                              void* d_out, int out_size, void* d_ws, size_t ws_size,
                              hipStream_t stream) {
  (void)in_sizes; (void)n_in; (void)out_size; (void)ws_size;
  const float* x   = (const float*)d_in[0];
  const float* pc  = (const float*)d_in[1];
  const float* Wq  = (const float*)d_in[2];
  const float* bq  = (const float*)d_in[3];
  const float* Wk  = (const float*)d_in[4];
  const float* bk  = (const float*)d_in[5];
  const float* Wv  = (const float*)d_in[6];
  const float* bv  = (const float*)d_in[7];
  const float* Wo  = (const float*)d_in[8];
  const float* bo  = (const float*)d_in[9];
  const float* pb  = (const float*)d_in[10];

  _Float16* ws = (_Float16*)d_ws;
  const size_t nX   = (size_t)B_ * N_ * D_;          // 2,097,152
  const size_t nW   = (size_t)D_ * D_;               //   262,144
  const size_t nAug = (size_t)B_ * H_ * N_ * AUG_;   // 3,145,728

  _Float16* xh   = ws;
  _Float16* Wqh  = xh  + nX;
  _Float16* Wkh  = Wqh + nW;
  _Float16* Wvh  = Wkh + nW;
  _Float16* Woh  = Wvh + nW;
  _Float16* Qaug = Woh + nW;
  _Float16* Kaug = Qaug + nAug;
  _Float16* Vt   = Kaug + nAug;                // [B*H, HD, N]
  _Float16* Ah   = Vt  + nX;                   // attended f16 [B*N, D]

  cvt_f32_f16<<<2048, 256, 0, stream>>>(x,  xh,  (int)nX);
  cvt_f32_f16<<<512,  256, 0, stream>>>(Wq, Wqh, (int)nW);
  cvt_f32_f16<<<512,  256, 0, stream>>>(Wk, Wkh, (int)nW);
  cvt_f32_f16<<<512,  256, 0, stream>>>(Wv, Wvh, (int)nW);
  cvt_f32_f16<<<512,  256, 0, stream>>>(Wo, Woh, (int)nW);

  proj_qkv<<<dim3(512, 1, 3), 128, 0, stream>>>(
      xh, Wqh, Wkh, Wvh, bq, bk, bv, Qaug, Kaug, Vt);

  phase_fill<<<(B_ * H_ * N_ + 255) / 256, 256, 0, stream>>>(pc, pb, Qaug, Kaug);

  flash_attn<<<B_ * H_ * (N_ / 128), 128, 0, stream>>>(Qaug, Kaug, Vt, Ah);

  gemm_out<<<512, 128, 0, stream>>>(Ah, Woh, bo, (float*)d_out);
}
